// DurationConditioningProjector_54949811585156
// MI455X (gfx1250) — compile-verified, matrix operand order checked by reading, not run on hardware
//
#include <hip/hip_runtime.h>
#include <cmath>

// ---- CDNA5 WMMA types -------------------------------------------------------
typedef __attribute__((ext_vector_type(16))) __bf16 v16bf;
typedef __attribute__((ext_vector_type(8)))  __bf16 v8bf;
typedef __attribute__((ext_vector_type(8)))  float  v8f;

__device__ __forceinline__ v8f wmma_bf16(v16bf a, v16bf b, v8f c) {
  // D = A(16x32 bf16) * B(32x16 bf16) + C(16x16 f32)
  return __builtin_amdgcn_wmma_f32_16x16x32_bf16(
      /*neg_a=*/false, a, /*neg_b=*/false, b,
      /*c_mod=*/(short)0, c, /*reuse_a=*/false, /*reuse_b=*/false);
}

// A-fragment (16x32, M=rows, K=cols) from a row-major bf16 tile in LDS.
// ISA layout: lane l holds M=l&15; elements e: K = (e>>3)*16 + (l>>4)*8 + (e&7)
// => two contiguous 8-element (16B) runs per lane.
__device__ __forceinline__ v16bf load_A_frag(const __bf16* tile, int stride,
                                             int m0, int k0) {
  int lane = threadIdx.x & 31;
  const __bf16* p = tile + (size_t)(m0 + (lane & 15)) * stride + k0 + ((lane >> 4) << 3);
  union { v16bf v; v8bf h[2]; } u;
  u.h[0] = *reinterpret_cast<const v8bf*>(p);
  u.h[1] = *reinterpret_cast<const v8bf*>(p + 16);
  return u.v;
}

// B-fragment pre-packed so each lane's 16 bf16 are contiguous (32B/lane).
__device__ __forceinline__ v16bf load_B_frag(const __bf16* packed, int frag) {
  int lane = threadIdx.x & 31;
  return *reinterpret_cast<const v16bf*>(packed + (size_t)frag * 512 + lane * 16);
}

// K-pattern inside a 32-wide chunk for lane/element (matches load_A_frag / B layout)
__device__ __forceinline__ int kpat(int e, int lane) {
  return ((e >> 3) << 4) + ((lane >> 4) << 3) + (e & 7);
}

// ---- weight packing ---------------------------------------------------------
// Generic W (NOUT x KDIM row-major) -> fragments [kc*NTILES + n][lane][e]
__global__ void pack_w_kernel(const float* __restrict__ W, __bf16* __restrict__ packed,
                              int KDIM, int NTILES) {
  int tid = blockIdx.x * 256 + threadIdx.x;
  int frags = (KDIM / 32) * NTILES;
  if (tid >= frags * 32) return;
  int frag = tid >> 5, lane = tid & 31;
  int kc = frag / NTILES, n = frag % NTILES;
  int co = n * 16 + (lane & 15);
  for (int e = 0; e < 16; ++e) {
    int kk = kc * 32 + kpat(e, lane);
    packed[(size_t)frag * 512 + lane * 16 + e] = (__bf16)W[(size_t)co * KDIM + kk];
  }
}

// conv_w (L=3, Cout=128, Cin=128, K=31) -> per-layer frags [(k*4+kc)*8 + n]
__global__ void pack_conv_kernel(const float* __restrict__ W, __bf16* __restrict__ packed) {
  int tid = blockIdx.x * 256 + threadIdx.x;
  if (tid >= 3 * 992 * 32) return;
  int lane = tid & 31;
  int frag = tid >> 5;          // 0..2975
  int layer = frag / 992;
  int f = frag % 992;
  int kct = f >> 3, n = f & 7;
  int k = kct >> 2, kc = kct & 3;
  int co = n * 16 + (lane & 15);
  for (int e = 0; e < 16; ++e) {
    int ci = kc * 32 + kpat(e, lane);
    packed[(size_t)frag * 512 + lane * 16 + e] =
        (__bf16)W[(((size_t)layer * 128 + co) * 128 + ci) * 31 + k];
  }
}

__global__ void tobf_kernel(const float* __restrict__ in, __bf16* __restrict__ out, int n) {
  int gid = blockIdx.x * 256 + threadIdx.x;
  if (gid < n) out[gid] = (__bf16)in[gid];
}

// ---- duration cumsum + searchsorted(right) ----------------------------------
__global__ void idx_kernel(const int* __restrict__ dur, int* __restrict__ gidx, int T) {
  __shared__ int cum[1024];
  int b = blockIdx.x, tid = threadIdx.x;
  for (int i = tid; i < 1024; i += 256) cum[i] = dur[b * 1024 + i];
  __syncthreads();
  if (tid == 0) { int s = 0; for (int i = 0; i < 1024; ++i) { s += cum[i]; cum[i] = s; } }
  __syncthreads();
  int total = cum[1023];
  for (int t = tid; t < T; t += 256) {
    int lo = 0, hi = 1024;
    while (lo < hi) { int mid = (lo + hi) >> 1; if (cum[mid] <= t) lo = mid + 1; else hi = mid; }
    if (lo > 1023) lo = 1023;
    gidx[(size_t)b * T + t] = (t < total) ? (b * 1024 + lo) : -1;
  }
}

// ---- sinusoidal positional features (bf16, 128-wide rows) -------------------
__global__ void emb_kernel(const float* __restrict__ rel, __bf16* __restrict__ out, int total) {
  int gid = blockIdx.x * 256 + threadIdx.x;
  if (gid >= total) return;
  int row = gid >> 7, j = gid & 127;
  int fi = j & 63;
  float freq = __expf(-9.210340371976184f * (float)fi * (1.0f / 63.0f));
  float a = rel[row] * freq;
  float v = (j < 64) ? __sinf(a) : __cosf(a);
  out[gid] = (__bf16)v;
}

// ---- LayerNorm (+ optional exact GELU), fp32 in -> bf16 out, wave per row ---
__global__ void ln_kernel(const float* __restrict__ x, const float* __restrict__ g,
                          const float* __restrict__ bta, __bf16* __restrict__ out,
                          int rows, int doGelu) {
  int tid = threadIdx.x;
  int lane = tid & 31, wave = tid >> 5;
  int row = blockIdx.x * 8 + wave;
  if (row >= rows) return;
  const float* xr = x + (size_t)row * 128;
  float v[4]; float s = 0.f, ss = 0.f;
  for (int j = 0; j < 4; ++j) { v[j] = xr[lane + 32 * j]; s += v[j]; ss += v[j] * v[j]; }
  for (int m = 16; m >= 1; m >>= 1) { s += __shfl_xor(s, m, 32); ss += __shfl_xor(ss, m, 32); }
  float mean = s * (1.f / 128.f);
  float var = ss * (1.f / 128.f) - mean * mean;
  float rs = rsqrtf(var + 1e-5f);
  for (int j = 0; j < 4; ++j) {
    int c = lane + 32 * j;
    float y = (v[j] - mean) * rs * g[c] + bta[c];
    if (doGelu) y = 0.5f * y * (1.f + erff(y * 0.7071067811865475f));
    out[(size_t)row * 128 + c] = (__bf16)y;
  }
}

// ---- generic row GEMM via WMMA: out(rows x NT*16) = A(rows x KDIM) @ Wpacked^T
// MODE 0: out = acc + bias            MODE 1: out = acc + bias + gather(ppool)
template<int KDIM, int NTILES, int MODE>
__global__ void gemm_kernel(const __bf16* __restrict__ A, const __bf16* __restrict__ wpack,
                            const float* __restrict__ bias, float* __restrict__ out,
                            int rows, const float* __restrict__ gsrc,
                            const int* __restrict__ gidx) {
  constexpr int SK = KDIM + 8;                 // +16B row pad: bank-conflict-free
  __shared__ __align__(16) __bf16 tile[32 * SK];
  int row0 = blockIdx.x * 32;
  int tid = threadIdx.x;
  constexpr int CH = KDIM / 8;                 // 16B chunks per row
  for (int c = tid; c < 32 * CH; c += 256) {
    int r = c / CH, ch = c % CH;
    int row = row0 + r;
    float4 v = make_float4(0.f, 0.f, 0.f, 0.f);
    if (row < rows) v = *reinterpret_cast<const float4*>(A + (size_t)row * KDIM + ch * 8);
    *reinterpret_cast<float4*>(tile + r * SK + ch * 8) = v;
  }
  __syncthreads();
  int lane = tid & 31, wave = tid >> 5;
  constexpr int NJ = NTILES / 8;
  v8f acc[2][NJ] = {};
  for (int kc = 0; kc < KDIM / 32; ++kc) {
    v16bf a0 = load_A_frag(tile, SK, 0, kc * 32);
    v16bf a1 = load_A_frag(tile, SK, 16, kc * 32);
    for (int nj = 0; nj < NJ; ++nj) {
      v16bf bf = load_B_frag(wpack, kc * NTILES + wave + nj * 8);
      acc[0][nj] = wmma_bf16(a0, bf, acc[0][nj]);
      acc[1][nj] = wmma_bf16(a1, bf, acc[1][nj]);
    }
  }
  int mb = (lane >> 4) << 3;
  for (int nj = 0; nj < NJ; ++nj) {
    int col = (wave + nj * 8) * 16 + (lane & 15);
    float bv = bias[col];
    for (int m = 0; m < 2; ++m) {
      for (int rr = 0; rr < 8; ++rr) {
        int row = row0 + m * 16 + mb + rr;
        if (row < rows) {
          float v = acc[m][nj][rr] + bv;
          if (MODE == 1) { int g = gidx[row]; if (g >= 0) v += gsrc[(size_t)g * 128 + col]; }
          out[(size_t)row * (NTILES * 16) + col] = v;
        }
      }
    }
  }
}

// ---- dilated causal conv as WMMA GEMM (K-dim = 31 taps x 128 Cin = 124 chunks)
// x[t] += sum_k sum_ci h[t - (30-k)*dil, ci] * W[co,ci,k] + bias[co]
__global__ void conv_kernel(const __bf16* __restrict__ h, float* __restrict__ x,
                            const __bf16* __restrict__ wpack, const float* __restrict__ bias,
                            int T, int dil) {
  __shared__ __align__(16) __bf16 tile[152 * 136];   // max rows: 32 + 30*4
  const int b = blockIdx.y;
  const int t0 = blockIdx.x * 32;
  const int tid = threadIdx.x;
  const int R = 32 + 30 * dil;
  for (int c = tid; c < R * 16; c += 256) {
    int r = c >> 4, ch = c & 15;
    int t = t0 - 30 * dil + r;
    float4 v = make_float4(0.f, 0.f, 0.f, 0.f);
    if (t >= 0 && t < T)
      v = *reinterpret_cast<const float4*>(h + ((size_t)b * T + t) * 128 + ch * 8);
    *reinterpret_cast<float4*>(tile + r * 136 + ch * 8) = v;
  }
  __syncthreads();
  const int lane = tid & 31, wave = tid >> 5;        // wave = Cout tile (8 tiles)
  v8f acc0 = {}, acc1 = {};
  for (int kct = 0; kct < 124; ++kct) {
    int k = kct >> 2, kc = kct & 3;
    v16bf a0 = load_A_frag(tile, 136, k * dil, kc * 32);
    v16bf a1 = load_A_frag(tile, 136, 16 + k * dil, kc * 32);
    v16bf bf = load_B_frag(wpack, kct * 8 + wave);
    acc0 = wmma_bf16(a0, bf, acc0);
    acc1 = wmma_bf16(a1, bf, acc1);
  }
  int col = wave * 16 + (lane & 15);
  int mb = (lane >> 4) << 3;
  float bv = bias[col];
  for (int rr = 0; rr < 8; ++rr) {
    int t = t0 + mb + rr;
    if (t < T) { size_t o = ((size_t)b * T + t) * 128 + col; x[o] += acc0[rr] + bv; }
    int t2 = t0 + 16 + mb + rr;
    if (t2 < T) { size_t o = ((size_t)b * T + t2) * 128 + col; x[o] += acc1[rr] + bv; }
  }
}

// ---- host launch ------------------------------------------------------------
extern "C" void kernel_launch(void* const* d_in, const int* in_sizes, int n_in,
                              void* d_out, int out_size, void* d_ws, size_t ws_size,
                              hipStream_t stream) {
  const float* pooled   = (const float*)d_in[0];
  const float* rel_pos  = (const float*)d_in[1];
  const float* in_w     = (const float*)d_in[2];
  const float* in_b     = (const float*)d_in[3];
  const float* pos_w    = (const float*)d_in[4];
  const float* pos_b    = (const float*)d_in[5];
  const float* conv_w   = (const float*)d_in[6];
  const float* conv_b   = (const float*)d_in[7];
  const float* ln_g     = (const float*)d_in[8];
  const float* ln_b     = (const float*)d_in[9];
  const float* out_ln_g = (const float*)d_in[10];
  const float* out_ln_b = (const float*)d_in[11];
  const float* out_w    = (const float*)d_in[12];
  const float* out_b    = (const float*)d_in[13];
  const int*   dur      = (const int*)d_in[14];
  float* out = (float*)d_out;

  const int B = 16, N = 1024, DIN = 256;
  const int T = in_sizes[1] / B;
  const int rowsBT = B * T;

  char* ws = (char*)d_ws;
  size_t off = 0;
  auto alloc = [&](size_t bytes) -> void* {
    void* p = ws + off; off = (off + bytes + 255) & ~(size_t)255; return p;
  };
  __bf16* pooled_bf = (__bf16*)alloc((size_t)B * N * DIN * 2);
  float*  ppool     = (float*) alloc((size_t)B * N * 128 * 4);
  __bf16* emb       = (__bf16*)alloc((size_t)rowsBT * 128 * 2);
  float*  x         = (float*) alloc((size_t)rowsBT * 128 * 4);
  __bf16* hbuf      = (__bf16*)alloc((size_t)rowsBT * 128 * 2);
  int*    gidx      = (int*)   alloc((size_t)rowsBT * 4);
  __bf16* wp_in     = (__bf16*)alloc((size_t)64 * 512 * 2);
  __bf16* wp_pos    = (__bf16*)alloc((size_t)32 * 512 * 2);
  __bf16* wp_out    = (__bf16*)alloc((size_t)64 * 512 * 2);
  __bf16* wp_conv   = (__bf16*)alloc((size_t)2976 * 512 * 2);

  // weight packing (every call; deterministic)
  pack_w_kernel<<<(64 * 32 + 255) / 256, 256, 0, stream>>>(in_w,  wp_in,  256, 8);
  pack_w_kernel<<<(32 * 32 + 255) / 256, 256, 0, stream>>>(pos_w, wp_pos, 128, 8);
  pack_w_kernel<<<(64 * 32 + 255) / 256, 256, 0, stream>>>(out_w, wp_out, 128, 16);
  pack_conv_kernel<<<(2976 * 32 + 255) / 256, 256, 0, stream>>>(conv_w, wp_conv);
  tobf_kernel<<<(B * N * DIN + 255) / 256, 256, 0, stream>>>(pooled, pooled_bf, B * N * DIN);

  // duration indices, pos features
  idx_kernel<<<B, 256, 0, stream>>>(dur, gidx, T);
  emb_kernel<<<(rowsBT * 128 + 255) / 256, 256, 0, stream>>>(rel_pos, emb, rowsBT * 128);

  // in-projection on pooled (gather commutes with row-wise matmul)
  gemm_kernel<256, 8, 0><<<(B * N + 31) / 32, 256, 0, stream>>>(
      pooled_bf, wp_in, in_b, ppool, B * N, nullptr, nullptr);

  // x = gather(ppool) + posemb @ pos_w^T + pos_b
  gemm_kernel<128, 8, 1><<<(rowsBT + 31) / 32, 256, 0, stream>>>(
      emb, wp_pos, pos_b, x, rowsBT, ppool, gidx);

  // 3 residual dilated-conv blocks
  for (int i = 0; i < 3; ++i) {
    ln_kernel<<<(rowsBT + 7) / 8, 256, 0, stream>>>(
        x, ln_g + i * 128, ln_b + i * 128, hbuf, rowsBT, 1);
    conv_kernel<<<dim3((T + 31) / 32, B), 256, 0, stream>>>(
        hbuf, x, wp_conv + (size_t)i * 992 * 512, conv_b + i * 128, T, 1 << i);
  }

  // final LN + out projection -> d_out (B,T,256)
  ln_kernel<<<(rowsBT + 7) / 8, 256, 0, stream>>>(x, out_ln_g, out_ln_b, hbuf, rowsBT, 0);
  gemm_kernel<128, 16, 0><<<(rowsBT + 31) / 32, 256, 0, stream>>>(
      hbuf, wp_out, out_b, out, rowsBT, nullptr, nullptr);
}
